// VAE_14482629722138
// MI455X (gfx1250) — matrix operand order repaired and checked by reading, not compiled, not run on hardware
//
#include <hip/hip_runtime.h>
#include <hip/hip_bf16.h>
#include <stddef.h>

typedef __attribute__((ext_vector_type(16))) _Float16 v16h;
typedef __attribute__((ext_vector_type(8)))  _Float16 v8h;
typedef __attribute__((ext_vector_type(8)))  float    v8f;

#define BSZ   512
#define TENC  64
#define TDEC  64
#define NACT  64
#define ACTE  64
#define CFD   512
#define ZDIM  128
#define TDIMC 520

__device__ __forceinline__ float sigmf(float x){ return 1.0f/(1.0f + __expf(-x)); }

// =============================================================================
// Guard-free f16 WMMA GEMM:  C[M,N] = A[M,K] * W[N,K]^T (+bias[N]) (+D[M,N])
// Requirements (enforced by host-side padding): M%64==0 via grid, N%64==0,
// K%32==0, lda/ldw multiples of 16 elements, buffers 256B-aligned.
// Block = 4 waves; each wave owns a 16x64 strip: one A fragment is reused
// against 4 B fragments per 32-wide K chunk (4 v_wmma per chunk).
// A-frag layout (16x32 f16): lane l row=l&15, halves 0..7 -> k0+8*(l>>4)+0..7,
// halves 8..15 -> +16. B-frag: lane l col=l&15, same k order as A-frag.
// =============================================================================
__global__ __launch_bounds__(128)
void gemm_wmma(const _Float16* __restrict__ A, int lda,
               const _Float16* __restrict__ W, int ldw,
               const float* __restrict__ bias,
               const float* __restrict__ D, int ldd,
               void* __restrict__ Cp, int ldc,
               int K, int relu, int half_out)
{
  const int lane = threadIdx.x & 31;
  const int wave = threadIdx.x >> 5;
  const int grp  = lane >> 4;
  const int l16  = lane & 15;
  const int row0 = (blockIdx.y * 4 + wave) * 16;
  const int col0 = blockIdx.x * 64;
  const int m    = row0 + l16;

  v8f acc[4] = {};
  const _Float16* ap  = A + (size_t)m * lda + 8*grp;
  const _Float16* bp0 = W + (size_t)(col0 + l16) * ldw + 8*grp;
  const size_t bstep = (size_t)16 * ldw;

  for (int k0 = 0; k0 < K; k0 += 32) {
    v8h a0 = *(const v8h*)(ap + k0);
    v8h a1 = *(const v8h*)(ap + k0 + 16);
    v16h a = __builtin_shufflevector(a0, a1, 0,1,2,3,4,5,6,7,8,9,10,11,12,13,14,15);
#pragma unroll
    for (int ct = 0; ct < 4; ++ct) {
      v8h b0 = *(const v8h*)(bp0 + (size_t)ct*bstep + k0);
      v8h b1 = *(const v8h*)(bp0 + (size_t)ct*bstep + k0 + 16);
      v16h b = __builtin_shufflevector(b0, b1, 0,1,2,3,4,5,6,7,8,9,10,11,12,13,14,15);
      acc[ct] = __builtin_amdgcn_wmma_f32_16x16x32_f16(false, a, false, b,
                                                       (short)0, acc[ct], false, false);
    }
  }

  // C/D f32 16x16 layout: vgpr j -> row row0 + j + 8*grp, col = tile base + l16
#pragma unroll
  for (int ct = 0; ct < 4; ++ct) {
    const int n = col0 + ct*16 + l16;
    const float bv = bias ? bias[n] : 0.0f;
#pragma unroll
    for (int j = 0; j < 8; ++j) {
      const int mr = row0 + j + 8*grp;
      float v = acc[ct][j] + bv;
      if (D)    v += D[(size_t)mr*ldd + n];
      if (relu) v = fmaxf(v, 0.0f);
      if (half_out) ((_Float16*)Cp)[(size_t)mr*ldc + n] = (_Float16)v;
      else          ((float*)Cp)   [(size_t)mr*ldc + n] = v;
    }
  }
}

// ------------------------------ small kernels --------------------------------
__global__ void zero_f(float* p, int n){
  for (int i = blockIdx.x*blockDim.x + threadIdx.x; i < n; i += gridDim.x*blockDim.x)
    p[i] = 0.0f;
}
__global__ void zero_h(_Float16* p, int n){
  for (int i = blockIdx.x*blockDim.x + threadIdx.x; i < n; i += gridDim.x*blockDim.x)
    p[i] = (_Float16)0.0f;
}
__global__ void copy_f(const float* __restrict__ s, float* __restrict__ d, int n){
  int i = blockIdx.x*blockDim.x + threadIdx.x;
  if (i < n) d[i] = s[i];
}
// f32 submatrix -> f16 submatrix (one-time weight conversion / repacking)
__global__ void conv_cols(const float* __restrict__ src, int lds, int scol,
                          _Float16* __restrict__ dst, int ldd, int dcol,
                          int rows, int cols){
  int idx = blockIdx.x*blockDim.x + threadIdx.x;
  if (idx >= rows*cols) return;
  int r = idx / cols, c = idx % cols;
  dst[(size_t)r*ldd + dcol + c] = (_Float16)src[(size_t)r*lds + scol + c];
}
__global__ void bias3_k(const float* a1, const float* a2, const float* b1, const float* b2,
                        const float* c1, const float* c2,
                        float* be, float* ba, float* bt){
  int n = blockIdx.x*blockDim.x + threadIdx.x;
  if (n >= 4*CFD) return;
  be[n] = a1[n] + a2[n];
  ba[n] = b1[n] + b2[n];
  bt[n] = c1[n] + c2[n];
}
__global__ void lens_k(const int* __restrict__ acts, int* __restrict__ lens){
  int b = blockIdx.x*blockDim.x + threadIdx.x;
  if (b >= BSZ) return;
  int bv = -1, bi = 0;
  for (int t = 0; t < TENC; ++t){ int v = acts[b*TENC + t]; if (v > bv){ bv = v; bi = t; } }
  lens[b] = bi;   // first pad position == sequence length
}
__global__ void eattrs_k(const int* __restrict__ cat, const float* __restrict__ num,
                         const float* __restrict__ emb, const float* __restrict__ W,
                         const float* __restrict__ bb, _Float16* __restrict__ tf, int ldt){
  int idx = blockIdx.x*blockDim.x + threadIdx.x;
  if (idx >= BSZ*8) return;
  int b = idx >> 3, o = idx & 7;
  int c = cat[b];
  float s = bb[o];
  for (int k = 0; k < 16; ++k) s += emb[c*16 + k] * W[o*17 + k];
  s += num[b] * W[o*17 + 16];
  tf[(size_t)b*ldt + o] = (_Float16)fmaxf(s, 0.0f);
}
__global__ void build_xe(const int* __restrict__ acts, const float* __restrict__ ts,
                         const float* __restrict__ act_emb, _Float16* __restrict__ xe,
                         int ldx, int t){
  int idx = blockIdx.x*blockDim.x + threadIdx.x;
  if (idx >= BSZ*65) return;
  int b = idx / 65, c = idx % 65;
  float v = (c < 64) ? act_emb[(size_t)acts[b*TENC + t]*ACTE + c] : ts[b*TENC + t];
  xe[(size_t)b*ldx + c] = (_Float16)v;
}
// gate order i,f,g,o over [B,4*CF]; h written as f16 into h1 (and optionally h2);
// encoder also stores h into the cf slot of t_full when t == lens[b]-1.
__global__ void lstm_pw(const float* __restrict__ g, float* __restrict__ c,
                        _Float16* __restrict__ h1, int ldh1,
                        _Float16* __restrict__ h2, int ldh2,
                        _Float16* __restrict__ cf, int ldcf,
                        const int* __restrict__ lens, int t){
  int idx = blockIdx.x*blockDim.x + threadIdx.x;
  if (idx >= BSZ*CFD) return;
  int b = idx >> 9, j = idx & (CFD-1);
  const float* gr = g + (size_t)b*4*CFD;
  float gi = gr[j], gf = gr[CFD + j], gg = gr[2*CFD + j], go = gr[3*CFD + j];
  float cn = sigmf(gf)*c[idx] + sigmf(gi)*tanhf(gg);
  float h  = sigmf(go)*tanhf(cn);
  c[idx] = cn;
  h1[(size_t)b*ldh1 + j] = (_Float16)h;
  if (h2) h2[(size_t)b*ldh2 + j] = (_Float16)h;
  if (cf && t == lens[b]-1) cf[(size_t)b*ldcf + 8 + j] = (_Float16)h;
}
__global__ void z_k(const float* __restrict__ mean, const float* __restrict__ var,
                    const float* __restrict__ eps, _Float16* __restrict__ z){
  int idx = blockIdx.x*blockDim.x + threadIdx.x;
  if (idx >= BSZ*ZDIM) return;
  z[idx] = (_Float16)(mean[idx] + var[idx]*eps[idx]);
}
__global__ void tc2_k(const float* __restrict__ y1, const float* __restrict__ W,
                      const float* __restrict__ bb, float* __restrict__ out){
  int b = blockIdx.x*blockDim.x + threadIdx.x;
  if (b >= BSZ) return;
  float l[10]; float m = -1e30f;
  for (int i = 0; i < 10; ++i){
    float s = bb[i];
    for (int k = 0; k < 260; ++k) s += y1[(size_t)b*320 + k] * W[i*260 + k];
    l[i] = s; m = fmaxf(m, s);
  }
  float ssum = 0.0f;
  for (int i = 0; i < 10; ++i){ l[i] = __expf(l[i]-m); ssum += l[i]; }
  float inv = 1.0f/ssum;
  for (int i = 0; i < 10; ++i) out[(size_t)b*10 + i] = l[i]*inv;
}
__global__ void tn2_k(const float* __restrict__ y1, const float* __restrict__ W,
                      const float* __restrict__ bb, float* __restrict__ out){
  int b = blockIdx.x*blockDim.x + threadIdx.x;
  if (b >= BSZ) return;
  float s = bb[0];
  for (int k = 0; k < 260; ++k) s += y1[(size_t)b*320 + k] * W[k];
  out[b] = sigmf(s);
}
__global__ void ae_fill(_Float16* __restrict__ xa, int ldx, const float* __restrict__ act_emb){
  int idx = blockIdx.x*blockDim.x + threadIdx.x;
  if (idx >= BSZ*ACTE) return;
  int b = idx >> 6, j = idx & 63;
  xa[(size_t)b*ldx + j] = (_Float16)act_emb[(size_t)(NACT-1)*ACTE + j];   // eot row
}
// greedy feedback: softmax(logits) -> acts_rec, argmax embedding -> ae slots of
// both decoder LSTM input buffers.
__global__ void dec_am(const float* __restrict__ logits, const float* __restrict__ act_emb,
                       _Float16* __restrict__ xa, int ldxa,
                       _Float16* __restrict__ xt, int ldxt,
                       float* __restrict__ outp, int t){
  int b = blockIdx.x*blockDim.x + threadIdx.x;
  if (b >= BSZ) return;
  const float* l = logits + (size_t)b*NACT;
  float bv = l[0]; int bi = 0;
  for (int k = 1; k < NACT; ++k){ float v = l[k]; if (v > bv){ bv = v; bi = k; } }
  float s = 0.0f;
  for (int k = 0; k < NACT; ++k) s += __expf(l[k]-bv);
  float inv = 1.0f/s;
  float* op = outp + ((size_t)b*TDEC + t)*NACT;
  for (int k = 0; k < NACT; ++k) op[k] = __expf(l[k]-bv)*inv;
  const float* ar = act_emb + (size_t)bi*ACTE;
  for (int k = 0; k < ACTE; ++k){
    _Float16 v = (_Float16)ar[k];
    xa[(size_t)b*ldxa + k] = v;
    xt[(size_t)b*ldxt + k] = v;
  }
}
__global__ void ts_fin(const float* __restrict__ y1, const float* __restrict__ w,
                       const float* __restrict__ bb, _Float16* __restrict__ xt, int ldxt,
                       float* __restrict__ outts, int t){
  int b = blockIdx.x*blockDim.x + threadIdx.x;
  if (b >= BSZ) return;
  float s = bb[0];
  for (int k = 0; k < 256; ++k) s += y1[(size_t)b*256 + k] * w[k];
  xt[(size_t)b*ldxt + 64] = (_Float16)s;   // tsc carry for next step
  outts[(size_t)b*TDEC + t] = s;
}

// ------------------------------- host side -----------------------------------
static inline void gemm(hipStream_t s, const _Float16* A, int lda, const _Float16* W, int ldw,
                        const float* bias, const float* D, int ldd,
                        void* C, int ldc, int M, int N, int K, int relu, int half_out){
  dim3 g(N / 64, M / 64);
  gemm_wmma<<<g, 128, 0, s>>>(A, lda, W, ldw, bias, D, ldd, C, ldc, K, relu, half_out);
}

extern "C" void kernel_launch(void* const* d_in, const int* in_sizes, int n_in,
                              void* d_out, int out_size, void* d_ws, size_t ws_size,
                              hipStream_t stream)
{
  (void)in_sizes; (void)n_in; (void)out_size; (void)ws_size;
  const int*   attr_cat = (const int*)  d_in[0];
  const float* attr_num = (const float*)d_in[1];
  const int*   acts     = (const int*)  d_in[2];
  const float* ts_in    = (const float*)d_in[3];
  const float* eps      = (const float*)d_in[4];
  const float* attr_emb = (const float*)d_in[5];
  const float* a2a_W    = (const float*)d_in[6];
  const float* a2a_b    = (const float*)d_in[7];
  const float* act_emb  = (const float*)d_in[8];
  const float* eWih = (const float*)d_in[9];   const float* eWhh = (const float*)d_in[10];
  const float* ebih = (const float*)d_in[11];  const float* ebhh = (const float*)d_in[12];
  const float* mean_W = (const float*)d_in[13]; const float* mean_b = (const float*)d_in[14];
  const float* var_W  = (const float*)d_in[15]; const float* var_b  = (const float*)d_in[16];
  const float* z2t_W  = (const float*)d_in[17]; const float* z2t_b  = (const float*)d_in[18];
  const float* tc1_W  = (const float*)d_in[19]; const float* tc1_b  = (const float*)d_in[20];
  const float* tc2_W  = (const float*)d_in[21]; const float* tc2_b  = (const float*)d_in[22];
  const float* tn1_W  = (const float*)d_in[23]; const float* tn1_b  = (const float*)d_in[24];
  const float* tn2_W  = (const float*)d_in[25]; const float* tn2_b  = (const float*)d_in[26];
  const float* aWih = (const float*)d_in[27]; const float* aWhh = (const float*)d_in[28];
  const float* abih = (const float*)d_in[29]; const float* abhh = (const float*)d_in[30];
  const float* tWih = (const float*)d_in[31]; const float* tWhh = (const float*)d_in[32];
  const float* tbih = (const float*)d_in[33]; const float* tbhh = (const float*)d_in[34];
  const float* e2act_W = (const float*)d_in[35]; const float* e2act_b = (const float*)d_in[36];
  const float* ts1_W = (const float*)d_in[37]; const float* ts1_b = (const float*)d_in[38];
  const float* ts2_W = (const float*)d_in[39]; const float* ts2_b = (const float*)d_in[40];

  // ---- workspace allocator (256B aligned blocks) ----
  char* base = (char*)d_ws;
  size_t off = 0;
  auto alloc = [&](size_t bytes){
    char* p = base + off;
    off = (off + bytes + 255) & ~(size_t)255;
    return (void*)p;
  };
  auto hbuf = [&](size_t elems){ return (_Float16*)alloc(elems*2); };
  auto fbuf = [&](size_t elems){ return (float*)alloc(elems*4); };

  // f16 weights, K padded to %32, N padded to %64 (zero fill)
  _Float16* WeH   = hbuf((size_t)2048*608);  // [eWih(65) | eWhh(512)] K 577->608
  _Float16* WaH   = hbuf((size_t)2048*576);  // [aWih[:,520:584] | aWhh] K 576
  _Float16* WtH   = hbuf((size_t)2048*608);  // [tWih[:,520:585] | tWhh] K 577->608
  _Float16* Wa0H  = hbuf((size_t)2048*544);  // aWih[:, :520]  K 520->544
  _Float16* Wt0H  = hbuf((size_t)2048*544);  // tWih[:, :520]
  _Float16* WmnH  = hbuf((size_t)128*544);   // mean_W
  _Float16* WvrH  = hbuf((size_t)128*544);   // var_W
  _Float16* Wz2H  = hbuf((size_t)576*128);   // z2t_W, N 520->576
  _Float16* Wc1H  = hbuf((size_t)320*544);   // tc1_W, N 260->320
  _Float16* Wn1H  = hbuf((size_t)320*544);   // tn1_W
  _Float16* We2H  = hbuf((size_t)64*512);    // e2act_W
  _Float16* Ws1H  = hbuf((size_t)256*512);   // ts1_W
  // f16 activations
  _Float16* XeH   = hbuf((size_t)BSZ*608);   // [x_t(65) | h_e(512) | pad]
  _Float16* XaH   = hbuf((size_t)BSZ*576);   // [ae(64)  | h_a(512)]
  _Float16* XtH   = hbuf((size_t)BSZ*608);   // [ae2(64) | tsc(1) | h_t(512) | pad]
  _Float16* TfH   = hbuf((size_t)BSZ*544);   // [e_attrs(8) | cf(512) | pad]
  _Float16* ZbH   = hbuf((size_t)BSZ*128);
  _Float16* TrH   = hbuf((size_t)BSZ*576);   // t_rec (cols 520..575 == 0)
  _Float16* HtH   = hbuf((size_t)BSZ*512);
  // f32 state / gate buffers
  float* biasE = fbuf(2048); float* biasA = fbuf(2048); float* biasT = fbuf(2048);
  float* bz2   = fbuf(576);  float* bc1   = fbuf(320);  float* bn1   = fbuf(320);
  float* Ce    = fbuf((size_t)BSZ*CFD);
  float* Ca    = fbuf((size_t)BSZ*CFD);
  float* Ct    = fbuf((size_t)BSZ*CFD);
  float* Gates = fbuf((size_t)BSZ*2048);
  float* Ga0   = fbuf((size_t)BSZ*2048);
  float* Gt0   = fbuf((size_t)BSZ*2048);
  float* Y1    = fbuf((size_t)BSZ*320);
  float* Y1t   = fbuf((size_t)BSZ*256);
  float* Logit = fbuf((size_t)BSZ*NACT);
  int*   Lens  = (int*)alloc(BSZ*4);

  // output layout (flat, return order)
  float* out    = (float*)d_out;
  float* o_cat  = out;                              // [512,10]
  float* o_num  = out + 5120;                       // [512]
  float* o_acts = out + 5632;                       // [512,64,64]
  float* o_ts   = o_acts + (size_t)BSZ*TDEC*NACT;   // [512,64]
  float* o_mean = o_ts + (size_t)BSZ*TDEC;          // [512,128]
  float* o_var  = o_mean + (size_t)BSZ*ZDIM;        // [512,128]

  auto cvt = [&](const float* s, int lds, int sc, _Float16* d, int ldd, int dc, int r, int c){
    int n = r*c;
    conv_cols<<<(n + 255)/256, 256, 0, stream>>>(s, lds, sc, d, ldd, dc, r, c);
  };
  auto zh = [&](_Float16* p, size_t n){ zero_h<<<1024, 256, 0, stream>>>(p, (int)n); };
  auto zf = [&](float* p, size_t n){ zero_f<<<1024, 256, 0, stream>>>(p, (int)n); };

  // ---- one-time weight prep: zero pads, then convert valid regions ----
  zh(WeH, (size_t)2048*608); zh(WtH, (size_t)2048*608);
  zh(Wa0H,(size_t)2048*544); zh(Wt0H,(size_t)2048*544);
  zh(WmnH,(size_t)128*544);  zh(WvrH,(size_t)128*544);
  zh(Wz2H,(size_t)576*128);  zh(Wc1H,(size_t)320*544); zh(Wn1H,(size_t)320*544);
  cvt(eWih, 65, 0,    WeH, 608, 0,  2048, 65);
  cvt(eWhh, 512, 0,   WeH, 608, 65, 2048, 512);
  cvt(aWih, 584, 520, WaH, 576, 0,  2048, 64);
  cvt(aWhh, 512, 0,   WaH, 576, 64, 2048, 512);
  cvt(tWih, 585, 520, WtH, 608, 0,  2048, 65);   // ae(64)+tsc(1) columns
  cvt(tWhh, 512, 0,   WtH, 608, 65, 2048, 512);
  cvt(aWih, 584, 0,   Wa0H, 544, 0, 2048, 520);
  cvt(tWih, 585, 0,   Wt0H, 544, 0, 2048, 520);
  cvt(mean_W, 520, 0, WmnH, 544, 0, 128, 520);
  cvt(var_W,  520, 0, WvrH, 544, 0, 128, 520);
  cvt(z2t_W,  128, 0, Wz2H, 128, 0, 520, 128);
  cvt(tc1_W,  520, 0, Wc1H, 544, 0, 260, 520);
  cvt(tn1_W,  520, 0, Wn1H, 544, 0, 260, 520);
  cvt(e2act_W,512, 0, We2H, 512, 0, 64, 512);
  cvt(ts1_W,  512, 0, Ws1H, 512, 0, 256, 512);
  bias3_k<<<8, 256, 0, stream>>>(ebih, ebhh, abih, abhh, tbih, tbhh, biasE, biasA, biasT);
  zf(bz2, 576); copy_f<<<3, 256, 0, stream>>>(z2t_b, bz2, 520);
  zf(bc1, 320); copy_f<<<2, 256, 0, stream>>>(tc1_b, bc1, 260);
  zf(bn1, 320); copy_f<<<2, 256, 0, stream>>>(tn1_b, bn1, 260);

  // ---- encoder ----
  lens_k<<<2, 256, 0, stream>>>(acts, Lens);
  zh(TfH, (size_t)BSZ*544);
  eattrs_k<<<16, 256, 0, stream>>>(attr_cat, attr_num, attr_emb, a2a_W, a2a_b, TfH, 544);
  zh(XeH, (size_t)BSZ*608);
  zf(Ce,  (size_t)BSZ*CFD);
  for (int t = 0; t < TENC; ++t){
    build_xe<<<(BSZ*65 + 255)/256, 256, 0, stream>>>(acts, ts_in, act_emb, XeH, 608, t);
    gemm(stream, XeH, 608, WeH, 608, biasE, nullptr, 0, Gates, 2048, BSZ, 2048, 608, 0, 0);
    lstm_pw<<<1024, 256, 0, stream>>>(Gates, Ce, XeH + 65, 608, nullptr, 0, TfH, 544, Lens, t);
  }
  gemm(stream, TfH, 544, WmnH, 544, mean_b, nullptr, 0, o_mean, 128, BSZ, 128, 544, 0, 0);
  gemm(stream, TfH, 544, WvrH, 544, var_b,  nullptr, 0, o_var,  128, BSZ, 128, 544, 0, 0);
  z_k<<<256, 256, 0, stream>>>(o_mean, o_var, eps, ZbH);

  // ---- decoder static parts ----
  gemm(stream, ZbH, 128, Wz2H, 128, bz2, nullptr, 0, TrH, 576, BSZ, 576, 128, 1, 1);
  gemm(stream, TrH, 576, Wc1H, 544, bc1, nullptr, 0, Y1, 320, BSZ, 320, 544, 1, 0);
  tc2_k<<<2, 256, 0, stream>>>(Y1, tc2_W, tc2_b, o_cat);
  gemm(stream, TrH, 576, Wn1H, 544, bn1, nullptr, 0, Y1, 320, BSZ, 320, 544, 1, 0);
  tn2_k<<<2, 256, 0, stream>>>(Y1, tn2_W, tn2_b, o_num);
  gemm(stream, TrH, 576, Wa0H, 544, biasA, nullptr, 0, Ga0, 2048, BSZ, 2048, 544, 0, 0);
  gemm(stream, TrH, 576, Wt0H, 544, biasT, nullptr, 0, Gt0, 2048, BSZ, 2048, 544, 0, 0);

  zh(XaH, (size_t)BSZ*576);
  zh(XtH, (size_t)BSZ*608);
  zf(Ca,  (size_t)BSZ*CFD);
  zf(Ct,  (size_t)BSZ*CFD);
  ae_fill<<<128, 256, 0, stream>>>(XaH, 576, act_emb);

  // ---- decoder recurrence ----
  for (int t = 0; t < TDEC; ++t){
    gemm(stream, XaH, 576, WaH, 576, nullptr, Ga0, 2048, Gates, 2048, BSZ, 2048, 576, 0, 0);
    lstm_pw<<<1024, 256, 0, stream>>>(Gates, Ca, XaH + 64, 576, nullptr, 0, nullptr, 0, nullptr, 0);
    gemm(stream, XaH + 64, 576, We2H, 512, e2act_b, nullptr, 0, Logit, 64, BSZ, 64, 512, 0, 0);
    dec_am<<<2, 256, 0, stream>>>(Logit, act_emb, XaH, 576, XtH, 608, o_acts, t);
    gemm(stream, XtH, 608, WtH, 608, nullptr, Gt0, 2048, Gates, 2048, BSZ, 2048, 608, 0, 0);
    lstm_pw<<<1024, 256, 0, stream>>>(Gates, Ct, XtH + 65, 608, HtH, 512, nullptr, 0, nullptr, 0);
    gemm(stream, HtH, 512, Ws1H, 512, ts1_b, nullptr, 0, Y1t, 256, BSZ, 256, 512, 1, 0);
    ts_fin<<<2, 256, 0, stream>>>(Y1t, ts2_W, ts2_b, XtH, 608, o_ts, t);
  }
}
